// SpatioTemporalAutoencoder_14405320311213
// MI455X (gfx1250) — matrix operand order repaired and checked by reading, not compiled, not run on hardware
//
#include <hip/hip_runtime.h>

// Problem constants (match reference)
#define T_  16
#define N_  10000
#define E_  320000
#define F_  128
#define H_  256
#define L_  64
#define HL_ 256
#define G4_ 1024      // 4*HL
#define NL_ 640000    // N*L

typedef __attribute__((ext_vector_type(16))) __bf16 v16bf;
typedef __attribute__((ext_vector_type(8)))  __bf16 v8bf;
typedef __attribute__((ext_vector_type(8)))  float  v8f;

static __device__ __forceinline__ __bf16 f2bf(float x) {
    union { float f; unsigned u; } v; v.f = x;
    unsigned r = v.u + 0x7FFFu + ((v.u >> 16) & 1u);   // round-to-nearest-even
    unsigned short h = (unsigned short)(r >> 16);
    __bf16 o; __builtin_memcpy(&o, &h, 2); return o;
}

// ---------------- utility ----------------
__global__ void zero_kernel(float* p, int n) {
    for (int i = blockIdx.x * blockDim.x + threadIdx.x; i < n;
         i += gridDim.x * blockDim.x) p[i] = 0.f;
}

// plain f32 -> bf16 conversion (activation matrices, layout preserved)
__global__ void cvt_bf16_kernel(const float* __restrict__ src, __bf16* __restrict__ dst, int n) {
    for (int i = blockIdx.x * blockDim.x + threadIdx.x; i < n;
         i += gridDim.x * blockDim.x) dst[i] = f2bf(src[i]);
}

// pack weight B[K,Nc] (row-major f32) into WMMA B-fragment layout:
// Bp[(((kt*Nc)+c)*2+h)*16 + i] = B[(kt*32 + h*16 + i)*Nc + c]
// -> each lane's 16 K-values for its column are one contiguous 32B chunk
__global__ void pack_b_kernel(const float* __restrict__ B, __bf16* __restrict__ Bp,
                              int K, int Nc) {
    int n = K * Nc;
    for (int idx = blockIdx.x * blockDim.x + threadIdx.x; idx < n;
         idx += gridDim.x * blockDim.x) {
        int k  = idx / Nc;
        int c  = idx - k * Nc;
        int kt = k >> 5, kr = k & 31;
        int h  = kr >> 4, i = kr & 15;
        Bp[(((size_t)kt * Nc + c) * 2 + h) * 16 + i] = f2bf(B[idx]);
    }
}

// deg[dst] += ew (or 1), plus self-loops with weight 1
__global__ void deg_kernel(const int* __restrict__ dst, const float* __restrict__ ew,
                           float* __restrict__ deg) {
    int n = E_ + N_;
    for (int i = blockIdx.x * blockDim.x + threadIdx.x; i < n;
         i += gridDim.x * blockDim.x) {
        if (i < E_) atomicAdd(&deg[dst[i]], ew ? ew[i] : 1.0f);
        else        atomicAdd(&deg[i - E_], 1.0f);
    }
}

__global__ void rsqrt_kernel(const float* __restrict__ deg, float* __restrict__ dis) {
    for (int i = blockIdx.x * blockDim.x + threadIdx.x; i < N_;
         i += gridDim.x * blockDim.x)
        dis[i] = deg[i] > 0.f ? rsqrtf(deg[i]) : 0.f;
}

// norm[e] = dis[src]*w*dis[dst]
__global__ void norm_kernel(const int* __restrict__ src, const int* __restrict__ dst,
                            const float* __restrict__ ew, const float* __restrict__ dis,
                            float* __restrict__ norm) {
    int n = E_ + N_;
    for (int i = blockIdx.x * blockDim.x + threadIdx.x; i < n;
         i += gridDim.x * blockDim.x) {
        int s, d; float w;
        if (i < E_) { s = src[i]; d = dst[i]; w = ew ? ew[i] : 1.0f; }
        else        { s = d = i - E_;          w = 1.0f; }
        norm[i] = dis[s] * w * dis[d];
    }
}

// ---------------- dense GEMM: C[M,Nc] = A[M,K] @ B[K,Nc] ----------------
// A: bf16 row-major [M,K]; Bp: pre-packed bf16 fragments; f32 accumulate.
// grid = (Nc/64, M/16), block = 128 (4 waves, one 16x16 tile each)
__global__ __launch_bounds__(128)
void gemm_wmma_kernel(const __bf16* __restrict__ A, const __bf16* __restrict__ Bp,
                      float* __restrict__ C, int M, int K, int Nc) {
    int lane = threadIdx.x & 31;
    int wave = threadIdx.x >> 5;
    int row0 = blockIdx.y * 16;
    int col0 = (blockIdx.x * 4 + wave) * 16;
    int half = lane >> 4;      // lane half selects K sub-ranges
    int lid  = lane & 15;
    int col  = col0 + lid;

    // A fragment: row row0+lid, chunks at K offsets half*16 and half*16+8
    const __bf16* ap = A + (size_t)(row0 + lid) * K + half * 16;
    // B fragment: packed, one 32B chunk per lane per k-tile
    const __bf16* bp = Bp + ((size_t)col * 2 + half) * 16;
    const size_t  bstep = (size_t)Nc * 32;

    v8f acc = {};
    int ktiles = K >> 5;
    #pragma unroll 2
    for (int kt = 0; kt < ktiles; ++kt) {
        v8bf a0 = *(const v8bf*)(ap);
        v8bf a1 = *(const v8bf*)(ap + 8);
        v16bf a = __builtin_shufflevector(a0, a1, 0, 1, 2, 3, 4, 5, 6, 7,
                                                  8, 9, 10, 11, 12, 13, 14, 15);
        v16bf b = *(const v16bf*)(bp);
        acc = __builtin_amdgcn_wmma_f32_16x16x32_bf16(
            false, a, false, b, (short)0, acc, false, false);
        ap += 32;
        bp += bstep;
    }
    // C 16x16 f32: element i -> row row0 + half*8 + i, col
    float* cp = C + (size_t)(row0 + half * 8) * Nc + col;
    #pragma unroll
    for (int i = 0; i < 8; ++i) cp[(size_t)i * Nc] = acc[i];
}

// ---------------- GCN scatter: out[dst] += norm[e] * h[src] ----------------
// grid = E_+N_, block = Fd
__global__ void scatter_kernel(const float* __restrict__ h, const float* __restrict__ norm,
                               const int* __restrict__ src, const int* __restrict__ dst,
                               int Fd, float* __restrict__ out) {
    int e = blockIdx.x;
    int f = threadIdx.x;
    int s, d;
    if (e < E_) { s = src[e]; d = dst[e]; }
    else        { s = d = e - E_; }
    float w = norm[e];
    atomicAdd(&out[(size_t)d * Fd + f], w * h[(size_t)s * Fd + f]);
}

__global__ void bias_act_kernel(float* __restrict__ x, const float* __restrict__ b,
                                int n, int mask, int relu) {
    for (int i = blockIdx.x * blockDim.x + threadIdx.x; i < n;
         i += gridDim.x * blockDim.x) {
        float v = x[i] + b[i & mask];
        x[i] = relu ? fmaxf(v, 0.f) : v;
    }
}

// column sums of [N_, H_] into out[H_]; block = 256, each block covers 128 rows
__global__ void colsum_kernel(const float* __restrict__ x, float* __restrict__ out) {
    int f  = threadIdx.x;
    int r0 = blockIdx.x * 128;
    int r1 = (r0 + 128 < N_) ? r0 + 128 : N_;
    float s = 0.f;
    for (int r = r0; r < r1; ++r) s += x[(size_t)r * H_ + f];
    atomicAdd(&out[f], s);
}

// zs = (sum/N) @ Wfce + bfce   (64 threads)
__global__ void fce_kernel(const float* __restrict__ pooled, const float* __restrict__ W,
                           const float* __restrict__ b, float* __restrict__ z) {
    int j = threadIdx.x;
    float s = 0.f;
    for (int k = 0; k < H_; ++k) s += pooled[k] * W[k * L_ + j];
    z[j] = s * (1.0f / (float)N_) + b[j];
}

// one LSTM step, single block of 1024 threads
__global__ __launch_bounds__(1024)
void lstm_step_kernel(const float* __restrict__ x, int xdim,
                      const float* __restrict__ Wih, const float* __restrict__ Whh,
                      const float* __restrict__ bih, const float* __restrict__ bhh,
                      float* __restrict__ h, float* __restrict__ c,
                      float* __restrict__ hout) {
    __shared__ float sx[HL_], sh[HL_], sg[G4_];
    int j = threadIdx.x;
    if (j < xdim) sx[j] = x[j];
    if (j < HL_)  sh[j] = h[j];
    __syncthreads();
    float g = bih[j] + bhh[j];
    for (int k = 0; k < xdim; ++k) g += sx[k] * Wih[k * G4_ + j];
    for (int k = 0; k < HL_;  ++k) g += sh[k] * Whh[k * G4_ + j];
    sg[j] = g;
    __syncthreads();
    if (j < HL_) {
        float ig = 1.f / (1.f + __expf(-sg[j]));
        float fg = 1.f / (1.f + __expf(-sg[HL_ + j]));
        float gg = tanhf(sg[2 * HL_ + j]);
        float og = 1.f / (1.f + __expf(-sg[3 * HL_ + j]));
        float cn = fg * c[j] + ig * gg;
        c[j] = cn;
        float hn = og * tanhf(cn);
        h[j] = hn;
        hout[j] = hn;
    }
}

// zlast = h_final @ Whead + bhead  (64 threads)
__global__ void head_kernel(const float* __restrict__ hfin, const float* __restrict__ W,
                            const float* __restrict__ b, float* __restrict__ z) {
    int j = threadIdx.x;
    float s = b[j];
    for (int k = 0; k < HL_; ++k) s += hfin[k] * W[k * L_ + j];
    z[j] = s;
}

// decoder fc: out[j] = relu(sum_k z[k]*Wfcd[k,j] + bfcd[j]), j < 640000
// grid = 2500 x 256 : streams the 164MB weight once, fully coalesced
__global__ void dec_fc_kernel(const float* __restrict__ z, const float* __restrict__ W,
                              const float* __restrict__ b, float* __restrict__ out) {
    __shared__ float sz[L_];
    if (threadIdx.x < L_) sz[threadIdx.x] = z[threadIdx.x];
    __syncthreads();
    size_t j = (size_t)blockIdx.x * blockDim.x + threadIdx.x;
    float s = b[j];
    #pragma unroll 8
    for (int k = 0; k < L_; ++k) s += sz[k] * W[(size_t)k * NL_ + j];
    out[j] = fmaxf(s, 0.f);
}

// ---------------- host orchestration ----------------
extern "C" void kernel_launch(void* const* d_in, const int* in_sizes, int n_in,
                              void* d_out, int out_size, void* d_ws, size_t ws_size,
                              hipStream_t stream) {
    (void)in_sizes; (void)n_in; (void)out_size; (void)ws_size;

    const float* xs    = (const float*)d_in[0];
    const float* eattr = (const float*)d_in[1];
    const float* We0   = (const float*)d_in[2];  const float* be0  = (const float*)d_in[3];
    const float* We1   = (const float*)d_in[4];  const float* be1  = (const float*)d_in[5];
    const float* Wfce  = (const float*)d_in[6];  const float* bfce = (const float*)d_in[7];
    const float* Wih0  = (const float*)d_in[8];  const float* Whh0 = (const float*)d_in[9];
    const float* bih0  = (const float*)d_in[10]; const float* bhh0 = (const float*)d_in[11];
    const float* Wih1  = (const float*)d_in[12]; const float* Whh1 = (const float*)d_in[13];
    const float* bih1  = (const float*)d_in[14]; const float* bhh1 = (const float*)d_in[15];
    const float* Whead = (const float*)d_in[16]; const float* bhead= (const float*)d_in[17];
    const float* Wfcd  = (const float*)d_in[18]; const float* bfcd = (const float*)d_in[19];
    const float* Wd0   = (const float*)d_in[20]; const float* bd0  = (const float*)d_in[21];
    const float* Wd1   = (const float*)d_in[22]; const float* bd1  = (const float*)d_in[23];
    const float* Wd2   = (const float*)d_in[24]; const float* bd2  = (const float*)d_in[25];
    const int*   ei    = (const int*)d_in[26];
    const int*   srcI  = ei;
    const int*   dstI  = ei + E_;
    float* out = (float*)d_out;

    // workspace layout (floats)
    float* ws       = (float*)d_ws;
    float* norm_enc = ws;                         // E+N
    float* norm_dec = norm_enc + (E_ + N_);       // E+N
    float* deg      = norm_dec + (E_ + N_);       // N
    float* dis      = deg + N_;                   // N
    float* bufH     = dis + N_;                   // N*H (GEMM out)
    float* bufA     = bufH + (size_t)N_ * H_;     // N*H
    float* bufB     = bufA + (size_t)N_ * H_;     // N*H
    float* pooled   = bufB + (size_t)N_ * H_;     // H
    float* zs       = pooled + H_;                // T*L
    float* h0       = zs + T_ * L_;               // HL
    float* c0       = h0 + HL_;                   // HL
    float* h1s      = c0 + HL_;                   // HL
    float* c1s      = h1s + HL_;                  // HL
    float* h1seq    = c1s + HL_;                  // T*HL
    float* h2tmp    = h1seq + T_ * HL_;           // HL
    float* zlast    = h2tmp + HL_;                // L
    float* xdec     = zlast + L_;                 // N*L
    // bf16 regions (sized in float units: bf16 elems / 2)
    __bf16* aBf   = (__bf16*)(xdec + NL_);        // M*K max = N*H  -> N*H/2 floats
    __bf16* We0p  = (__bf16*)((float*)aBf + ((size_t)N_ * H_) / 2);   // F*H
    __bf16* We1p  = (__bf16*)((float*)We0p + (F_ * H_) / 2);          // H*H
    __bf16* Wd0p  = (__bf16*)((float*)We1p + (H_ * H_) / 2);          // L*H
    __bf16* Wd1p  = (__bf16*)((float*)Wd0p + (L_ * H_) / 2);          // H*H
    __bf16* Wd2p  = (__bf16*)((float*)Wd1p + (H_ * H_) / 2);          // H*F

    auto Z = [&](float* p, int n) {
        zero_kernel<<<dim3(256), dim3(256), 0, stream>>>(p, n);
    };
    auto CVT = [&](const float* s, __bf16* d, int n) {
        cvt_bf16_kernel<<<dim3(512), dim3(256), 0, stream>>>(s, d, n);
    };
    const dim3 gs(512), bs(256);

    // ---- one-time weight packing into WMMA B-fragment layout ----
    pack_b_kernel<<<gs, bs, 0, stream>>>(We0, We0p, F_, H_);
    pack_b_kernel<<<gs, bs, 0, stream>>>(We1, We1p, H_, H_);
    pack_b_kernel<<<gs, bs, 0, stream>>>(Wd0, Wd0p, L_, H_);
    pack_b_kernel<<<gs, bs, 0, stream>>>(Wd1, Wd1p, H_, H_);
    pack_b_kernel<<<gs, bs, 0, stream>>>(Wd2, Wd2p, H_, F_);

    // ---- graph normalization (encoder: edge_attr weights; decoder: all-ones) ----
    Z(deg, N_);
    deg_kernel<<<gs, bs, 0, stream>>>(dstI, eattr, deg);
    rsqrt_kernel<<<dim3(64), bs, 0, stream>>>(deg, dis);
    norm_kernel<<<gs, bs, 0, stream>>>(srcI, dstI, eattr, dis, norm_enc);

    Z(deg, N_);
    deg_kernel<<<gs, bs, 0, stream>>>(dstI, nullptr, deg);
    rsqrt_kernel<<<dim3(64), bs, 0, stream>>>(deg, dis);
    norm_kernel<<<gs, bs, 0, stream>>>(srcI, dstI, nullptr, dis, norm_dec);

    // ---- encoder: per timestep GCN(F->H) relu, GCN(H->H) relu, mean pool, fc ----
    for (int t = 0; t < T_; ++t) {
        const float* xt = xs + (size_t)t * N_ * F_;
        CVT(xt, aBf, N_ * F_);
        gemm_wmma_kernel<<<dim3(H_ / 64, N_ / 16), 128, 0, stream>>>(aBf, We0p, bufH, N_, F_, H_);
        Z(bufA, N_ * H_);
        scatter_kernel<<<dim3(E_ + N_), dim3(H_), 0, stream>>>(bufH, norm_enc, srcI, dstI, H_, bufA);
        bias_act_kernel<<<gs, bs, 0, stream>>>(bufA, be0, N_ * H_, H_ - 1, 1);

        CVT(bufA, aBf, N_ * H_);
        gemm_wmma_kernel<<<dim3(H_ / 64, N_ / 16), 128, 0, stream>>>(aBf, We1p, bufH, N_, H_, H_);
        Z(bufB, N_ * H_);
        scatter_kernel<<<dim3(E_ + N_), dim3(H_), 0, stream>>>(bufH, norm_enc, srcI, dstI, H_, bufB);
        bias_act_kernel<<<gs, bs, 0, stream>>>(bufB, be1, N_ * H_, H_ - 1, 1);

        Z(pooled, H_);
        colsum_kernel<<<dim3((N_ + 127) / 128), dim3(H_), 0, stream>>>(bufB, pooled);
        fce_kernel<<<dim3(1), dim3(L_), 0, stream>>>(pooled, Wfce, bfce, zs + t * L_);
    }

    // ---- 2-layer LSTM over T steps (sequential, latency-bound) ----
    Z(h0, 4 * HL_);   // zeroes h0,c0,h1s,c1s (contiguous)
    for (int t = 0; t < T_; ++t)
        lstm_step_kernel<<<dim3(1), dim3(G4_), 0, stream>>>(
            zs + t * L_, L_, Wih0, Whh0, bih0, bhh0, h0, c0, h1seq + t * HL_);
    for (int t = 0; t < T_; ++t)
        lstm_step_kernel<<<dim3(1), dim3(G4_), 0, stream>>>(
            h1seq + t * HL_, HL_, Wih1, Whh1, bih1, bhh1, h1s, c1s, h2tmp);
    head_kernel<<<dim3(1), dim3(L_), 0, stream>>>(h1s, Whead, bhead, zlast);

    // ---- decoder ----
    dec_fc_kernel<<<dim3(NL_ / 256), dim3(256), 0, stream>>>(zlast, Wfcd, bfcd, xdec);

    CVT(xdec, aBf, N_ * L_);
    gemm_wmma_kernel<<<dim3(H_ / 64, N_ / 16), 128, 0, stream>>>(aBf, Wd0p, bufH, N_, L_, H_);
    Z(bufA, N_ * H_);
    scatter_kernel<<<dim3(E_ + N_), dim3(H_), 0, stream>>>(bufH, norm_dec, srcI, dstI, H_, bufA);
    bias_act_kernel<<<gs, bs, 0, stream>>>(bufA, bd0, N_ * H_, H_ - 1, 1);

    CVT(bufA, aBf, N_ * H_);
    gemm_wmma_kernel<<<dim3(H_ / 64, N_ / 16), 128, 0, stream>>>(aBf, Wd1p, bufH, N_, H_, H_);
    Z(bufB, N_ * H_);
    scatter_kernel<<<dim3(E_ + N_), dim3(H_), 0, stream>>>(bufH, norm_dec, srcI, dstI, H_, bufB);
    bias_act_kernel<<<gs, bs, 0, stream>>>(bufB, bd1, N_ * H_, H_ - 1, 1);

    CVT(bufB, aBf, N_ * H_);
    gemm_wmma_kernel<<<dim3(F_ / 64, N_ / 16), 128, 0, stream>>>(aBf, Wd2p, bufH, N_, H_, F_);
    Z(out, N_ * F_);
    scatter_kernel<<<dim3(E_ + N_), dim3(F_), 0, stream>>>(bufH, norm_dec, srcI, dstI, F_, out);
    bias_act_kernel<<<gs, bs, 0, stream>>>(out, bd2, N_ * F_, F_ - 1, 0);
}
